// RandomKNeighborsMHA_73650099191880
// MI455X (gfx1250) — matrix operand order repaired and checked by reading, not compiled, not run on hardware
//
#include <hip/hip_runtime.h>

// ---------------------------------------------------------------------------
// RandomKNeighborsMHA for MI455X (gfx1250, wave32, WMMA)
// B=2, L=2048, C=1024, H=16, K=32, Dh=64
//
// Roofline: 4 x (4096x1024x1024) GEMMs = 34.4 GFLOP vs ~120MB traffic
// (~5us @ 23.3 TB/s, fits in 192MB L2) -> compute bound -> push projections
// through v_wmma_f32_16x16x32_bf16 (f32 accumulate).
// GEMM: 128x128 block tile, BK=64 (16 WMMA between barriers), 2-deep LDS
// double buffer with 72-element padded pitch (bank-conflict-free b128 frag
// loads), peeled straight-line main loop. y = x @ W^T means the WMMA
// B-operand B[k][n] = W[n][k] is W's native row-major layout: no transpose.
// ---------------------------------------------------------------------------

typedef __bf16 bf16_t;
typedef __attribute__((ext_vector_type(16))) __bf16 v16bf;
typedef __attribute__((ext_vector_type(8)))  float  v8f;

#define Bq 2
#define Lq 2048
#define Cq 1024
#define Hq 16
#define Kn 32
#define DhQ 64

#define PITCH 72  // LDS row pitch in bf16 (144B = 36 dwords, 16B aligned)

__device__ inline unsigned bf_bits(float f) {
  unsigned u = __float_as_uint(f);
  return (u + 0x7FFFu + ((u >> 16) & 1u)) >> 16;  // RNE f32 -> bf16 bits
}
__device__ inline bf16_t f2bf(float f) {
  unsigned short h = (unsigned short)bf_bits(f);
  bf16_t r;
  __builtin_memcpy(&r, &h, sizeof(r));
  return r;
}
__device__ inline unsigned pack2bf(float lo, float hi) {
  return bf_bits(lo) | (bf_bits(hi) << 16);
}

// ------------------------------------------------------- vectorized convert
__global__ __launch_bounds__(256) void cvt_f32_bf16_x4(
    const float4* __restrict__ in, uint2* __restrict__ out, int n4) {
  int i = blockIdx.x * blockDim.x + threadIdx.x;
  if (i < n4) {
    float4 f = in[i];
    uint2 o;
    o.x = pack2bf(f.x, f.y);
    o.y = pack2bf(f.z, f.w);
    out[i] = o;
  }
}

// --------------------------------------------------------------- WMMA GEMM
union FragU {
  v16bf v;
  uint4 u[2];
};

__global__ __launch_bounds__(256) void gemm_bf16_wmma(
    const bf16_t* __restrict__ A,   // M x Kd row-major
    const bf16_t* __restrict__ Wb,  // N x Kd row-major (= B^T natively)
    float* __restrict__ Cmat,       // M x N row-major
    int M, int N, int Kd, float alpha) {
  __shared__ __align__(16) bf16_t sA[2][128 * PITCH];  // [m][k], padded
  __shared__ __align__(16) bf16_t sB[2][128 * PITCH];  // [n][k], padded

  const int n0   = blockIdx.x * 128;
  const int m0   = blockIdx.y * 128;
  const int tid  = threadIdx.x;
  const int lane = tid & 31;
  const int wave = tid >> 5;
  const int wm   = wave & 3;   // 4 slices of 32 rows (M)
  const int wn   = wave >> 2;  // 2 slices of 64 cols (N)
  const int l16  = lane & 15;
  const bool hi  = lane >= 16;

  v8f acc[2][4] = {};

  // staging: 128 rows x 2 halves of 32 bf16 (64B) per matrix per thread
  const int r  = tid >> 1;
  const int hf = tid & 1;
  const bf16_t* gA = A  + (size_t)(m0 + r) * Kd + hf * 32;
  const bf16_t* gB = Wb + (size_t)(n0 + r) * Kd + hf * 32;
  bf16_t* lA = &sA[0][r * PITCH + hf * 32];
  bf16_t* lB = &sB[0][r * PITCH + hf * 32];
  const int bufStride = 128 * PITCH;

  auto stage = [&](int buf, int kk) {
    const uint4* pa = (const uint4*)(gA + kk);
    const uint4* pb = (const uint4*)(gB + kk);
    uint4 a0 = pa[0], a1 = pa[1], a2 = pa[2], a3 = pa[3];
    uint4 b0 = pb[0], b1 = pb[1], b2 = pb[2], b3 = pb[3];
    uint4* da = (uint4*)(lA + buf * bufStride);
    uint4* db = (uint4*)(lB + buf * bufStride);
    da[0] = a0; da[1] = a1; da[2] = a2; da[3] = a3;
    db[0] = b0; db[1] = b1; db[2] = b2; db[3] = b3;
  };

  const int ka0 = hi ? 8 : 0;   // A: elems 0-7 -> K=ka0+e; 8-15 -> K=ka0+16+(e-8)
  const int kb0 = hi ? 8 : 0;   // B: elems 0-7 -> K=kb0+e; 8-15 -> K=kb0+16+(e-8)

  auto compute = [&](int cur) {
#pragma unroll
    for (int ksub = 0; ksub < 64; ksub += 32) {
      FragU afrag[2];
#pragma unroll
      for (int s = 0; s < 2; ++s) {
        const bf16_t* p = &sA[cur][(wm * 32 + s * 16 + l16) * PITCH + ksub];
        afrag[s].u[0] = *(const uint4*)(p + ka0);
        afrag[s].u[1] = *(const uint4*)(p + ka0 + 16);
      }
      FragU bfrag[4];
#pragma unroll
      for (int t = 0; t < 4; ++t) {
        const bf16_t* p = &sB[cur][(wn * 64 + t * 16 + l16) * PITCH + ksub];
        bfrag[t].u[0] = *(const uint4*)(p + kb0);
        bfrag[t].u[1] = *(const uint4*)(p + kb0 + 16);
      }
#pragma unroll
      for (int s = 0; s < 2; ++s)
#pragma unroll
        for (int t = 0; t < 4; ++t)
          acc[s][t] = __builtin_amdgcn_wmma_f32_16x16x32_bf16(
              false, afrag[s].v, false, bfrag[t].v, (short)0, acc[s][t],
              false, false);
    }
  };

  const int nsteps = Kd >> 6;  // K-step 64
  stage(0, 0);
  __syncthreads();

  for (int step = 0; step < nsteps - 1; ++step) {
    stage((step + 1) & 1, (step + 1) << 6);
    {  // pull tile s+2 toward near caches (clamped, prefetch is speculative)
      int kpf = (step + 2) < nsteps ? (step + 2) << 6 : (nsteps - 1) << 6;
      __builtin_prefetch(gA + kpf, 0, 3);
      __builtin_prefetch(gB + kpf, 0, 3);
    }
    compute(step & 1);
    __syncthreads();
  }
  compute((nsteps - 1) & 1);

  // ---- epilogue: C/D layout: VGPR j -> M=j+(hi?8:0), N=l%16
  const int mrowBase = m0 + wm * 32 + (hi ? 8 : 0);
  const int ncolBase = n0 + wn * 64 + l16;
#pragma unroll
  for (int s = 0; s < 2; ++s)
#pragma unroll
    for (int t = 0; t < 4; ++t)
#pragma unroll
      for (int j = 0; j < 8; ++j)
        Cmat[(size_t)(mrowBase + s * 16 + j) * N + ncolBase + t * 16] =
            alpha * acc[s][t][j];
}

// ----------------------------------------------------------- attention core
// One block per (b,l): 16 waves = 16 heads; wave32 lanes = the K=32 neighbors.
__global__ __launch_bounds__(512) void attn_kernel(
    const float* __restrict__ q, const float* __restrict__ k,
    const float* __restrict__ v, const int* __restrict__ idx,
    bf16_t* __restrict__ outb) {
  const int bl   = blockIdx.x;  // b*L + l
  const int b    = bl >> 11;    // / 2048
  const int l    = bl & (Lq - 1);
  const int h    = threadIdx.x >> 5;  // wave == head
  const int lane = threadIdx.x & 31;

  __shared__ int sidx[Kn];
  if (threadIdx.x < Kn) sidx[threadIdx.x] = idx[(size_t)l * Kn + threadIdx.x];
  __syncthreads();

  const int j = sidx[lane];
  const float* qrow = q + (size_t)bl * Cq + h * DhQ;            // wave-uniform
  const float* krow = k + (size_t)(b * Lq + j) * Cq + h * DhQ;  // per-lane

  // scores (q already pre-scaled by 1/sqrt(Dh) in the Q GEMM)
  float s = 0.f;
#pragma unroll 8
  for (int d = 0; d < DhQ; ++d) s += qrow[d] * krow[d];

  // wave32 softmax across the 32 neighbors
  float m = s;
#pragma unroll
  for (int off = 16; off >= 1; off >>= 1) m = fmaxf(m, __shfl_xor(m, off, 32));
  float e = __expf(s - m);
  float sum = e;
#pragma unroll
  for (int off = 16; off >= 1; off >>= 1) sum += __shfl_xor(sum, off, 32);
  const float p = e / sum;

  // weighted sum of V: lanes switch to owning 2 of the 64 dims
  const int d0 = lane * 2;
  float ax = 0.f, ay = 0.f;
#pragma unroll 4
  for (int kkk = 0; kkk < Kn; ++kkk) {
    float pk = __shfl(p, kkk, 32);
    int   jk = __shfl(j, kkk, 32);
    const float* vrow = v + (size_t)(b * Lq + jk) * Cq + h * DhQ + d0;
    ax += pk * vrow[0];
    ay += pk * vrow[1];
  }
  bf16_t* o = outb + (size_t)bl * Cq + h * DhQ + d0;
  o[0] = f2bf(ax);
  o[1] = f2bf(ay);
}

// ---------------------------------------------------------------- launcher
extern "C" void kernel_launch(void* const* d_in, const int* in_sizes, int n_in,
                              void* d_out, int out_size, void* d_ws,
                              size_t ws_size, hipStream_t stream) {
  const float* x   = (const float*)d_in[0];
  const int*   idx = (const int*)d_in[1];
  const float* Wq  = (const float*)d_in[2];
  const float* Wk  = (const float*)d_in[3];
  const float* Wv  = (const float*)d_in[4];
  const float* Wo  = (const float*)d_in[5];
  float*       out = (float*)d_out;

  char* ws = (char*)d_ws;
  const size_t MB = 1ull << 20;
  bf16_t* xb   = (bf16_t*)(ws + 0);        // 8 MiB  x in bf16
  bf16_t* Wqb  = (bf16_t*)(ws + 8 * MB);   // 2 MiB each, native [n][k] layout
  bf16_t* Wkb  = (bf16_t*)(ws + 10 * MB);
  bf16_t* Wvb  = (bf16_t*)(ws + 12 * MB);
  bf16_t* Wob  = (bf16_t*)(ws + 14 * MB);
  float*  qbuf = (float*)(ws + 16 * MB);   // 16 MiB each
  float*  kbuf = (float*)(ws + 32 * MB);
  float*  vbuf = (float*)(ws + 48 * MB);
  bf16_t* ab   = (bf16_t*)(ws + 64 * MB);  // 8 MiB attn output in bf16

  const int Mtot = Bq * Lq;            // 4096
  const int nx4  = Mtot * Cq / 4;      // x: 1 Mi float4
  const int nw4  = Cq * Cq / 4;        // W: 256 Ki float4

  cvt_f32_bf16_x4<<<(nx4 + 255) / 256, 256, 0, stream>>>(
      (const float4*)x, (uint2*)xb, nx4);
  cvt_f32_bf16_x4<<<(nw4 + 255) / 256, 256, 0, stream>>>(
      (const float4*)Wq, (uint2*)Wqb, nw4);
  cvt_f32_bf16_x4<<<(nw4 + 255) / 256, 256, 0, stream>>>(
      (const float4*)Wk, (uint2*)Wkb, nw4);
  cvt_f32_bf16_x4<<<(nw4 + 255) / 256, 256, 0, stream>>>(
      (const float4*)Wv, (uint2*)Wvb, nw4);
  cvt_f32_bf16_x4<<<(nw4 + 255) / 256, 256, 0, stream>>>(
      (const float4*)Wo, (uint2*)Wob, nw4);

  dim3 gg(Cq / 128, Mtot / 128);  // (8, 32)
  // q scaled by 1/sqrt(Dh)=0.125 inside the GEMM epilogue
  gemm_bf16_wmma<<<gg, 256, 0, stream>>>(xb, Wqb, qbuf, Mtot, Cq, Cq, 0.125f);
  gemm_bf16_wmma<<<gg, 256, 0, stream>>>(xb, Wkb, kbuf, Mtot, Cq, Cq, 1.0f);
  gemm_bf16_wmma<<<gg, 256, 0, stream>>>(xb, Wvb, vbuf, Mtot, Cq, Cq, 1.0f);

  attn_kernel<<<Bq * Lq, 512, 0, stream>>>(qbuf, kbuf, vbuf, idx, ab);

  gemm_bf16_wmma<<<gg, 256, 0, stream>>>(ab, Wob, out, Mtot, Cq, Cq, 1.0f);
}
